// AttentionCritic_16698832847183
// MI455X (gfx1250) — matrix operand order repaired and checked by reading, not compiled, not run on hardware
//
#include <hip/hip_runtime.h>

typedef unsigned short u16;
typedef __attribute__((ext_vector_type(16))) __bf16 v16bf;
typedef __attribute__((ext_vector_type(8)))  float  v8f;

namespace {
constexpr int kN = 16;      // agents
constexpr int kB = 4096;    // batch
constexpr int kS = 128;     // SDIM
constexpr int kA = 8;       // ADIM
constexpr int kI = 136;     // IDIM
constexpr int kIP = 160;    // IDIM padded to mult of 32
constexpr int kH = 256;     // hidden
constexpr int kHeads = 4;
constexpr int kD = 64;      // head dim
constexpr int kH2 = 512;    // 2*H
}

union V16 { v16bf v; uint4 q[2]; };
union Q4H { uint4 q; u16 h[8]; };

// hardware f32 -> bf16 (RNE) via native convert
__device__ __forceinline__ u16 f2bf(float f) {
  union { __bf16 b; u16 u; } c; c.b = (__bf16)f; return c.u;
}
__device__ __forceinline__ float lrelu(float x) { return x > 0.f ? x : 0.01f * x; }
__device__ __forceinline__ v8f vzero() { v8f z; for (int i = 0; i < 8; i++) z[i] = 0.f; return z; }

// A-matrix (16xK tile, bf16, row-major source with row stride `stride` halves).
// lane m=l%16; hi=0 lanes hold k0..k0+15,
// hi=1 lanes hold k0+16..k0+31 (contiguous halves).
__device__ __forceinline__ V16 loadA(const u16* base, size_t stride, int k0, int lane) {
  int m = lane & 15, hi = lane >> 4;
  const u16* p = base + (size_t)m * stride + k0 + hi * 16;
  V16 r; r.q[0] = *(const uint4*)p; r.q[1] = *(const uint4*)(p + 8); return r;
}
// B-matrix (Kx16 tile) sourced from a transposed [n][K] buffer (row stride `stride`).
// lanes 0-15: n=l, k=k0..k0+15 ; lanes 16-31: n=l-16, k=k0+16..k0+31 (contiguous halves).
__device__ __forceinline__ V16 loadB(const u16* base, size_t stride, int k0, int lane) {
  int n = lane & 15, hi = lane >> 4;
  const u16* p = base + (size_t)n * stride + k0 + hi * 16;
  V16 r; r.q[0] = *(const uint4*)p; r.q[1] = *(const uint4*)(p + 8); return r;
}
// speculative prefetch of the next B K-chunk (lowers to global_prefetch_b8)
__device__ __forceinline__ void pfB(const u16* base, size_t stride, int k0, int lane) {
  __builtin_prefetch((const void*)(base + (size_t)(lane & 15) * stride + k0 + (lane >> 4) * 16), 0, 0);
}
__device__ __forceinline__ v8f bwmma(V16 a, V16 b, v8f c) {
  return __builtin_amdgcn_wmma_f32_16x16x32_bf16(false, a.v, false, b.v, (short)0, c, false, false);
}

// ---------------- prep: weight transpose/convert f32[K][Nc] -> bf16[Npad][Kpad] ----------------
__global__ void wprep(const float* __restrict__ src, u16* __restrict__ dst,
                      int nmat, int K, int Nc, int Kpad, int Npad) {
  size_t total = (size_t)nmat * Npad * Kpad;
  size_t gs = (size_t)gridDim.x * blockDim.x;
  for (size_t idx = (size_t)blockIdx.x * blockDim.x + threadIdx.x; idx < total; idx += gs) {
    int k = (int)(idx % Kpad);
    size_t t = idx / Kpad;
    int nn = (int)(t % Npad);
    int m  = (int)(t / Npad);
    float v = (k < K && nn < Nc) ? src[((size_t)m * K + k) * Nc + nn] : 0.f;
    dst[idx] = f2bf(v);
  }
}

// ---------------- prep: activations f32 -> bf16 (sa padded 136->160) + zero reg accum ----------
__global__ void aprep(const float* __restrict__ sa, u16* __restrict__ saP,
                      const float* __restrict__ st, u16* __restrict__ stP,
                      float* __restrict__ regAcc) {
  size_t gid = (size_t)blockIdx.x * blockDim.x + threadIdx.x;
  size_t gs  = (size_t)gridDim.x * blockDim.x;
  if (gid < (size_t)kN) regAcc[gid] = 0.f;
  size_t tot1 = (size_t)kN * kB * kIP;
  for (size_t i = gid; i < tot1; i += gs) {
    int k = (int)(i % kIP);
    size_t row = i / kIP;
    saP[i] = (k < kI) ? f2bf(sa[row * kI + k]) : (u16)0;
  }
  size_t tot2 = (size_t)kN * kB * kS;
  for (size_t i = gid; i < tot2; i += gs) stP[i] = f2bf(st[i]);
}

// ------- encoders (M=32 register blocking: 2 row tiles share every B load) -------------------
__global__ __launch_bounds__(256) void enc_kernel(
    const u16* __restrict__ saP, const u16* __restrict__ stP,
    const u16* __restrict__ WencT, const float* __restrict__ b_enc,
    const u16* __restrict__ WsencT, const float* __restrict__ b_senc,
    u16* __restrict__ saEnc, u16* __restrict__ sEnc) {
  int a = blockIdx.y, b0 = blockIdx.x * 32;
  int wave = threadIdx.x >> 5, lane = threadIdx.x & 31;
  int n = lane & 15, hi = lane >> 4;
  const u16* Asa0 = saP + (size_t)(a * kB + b0) * kIP;
  const u16* Asa1 = Asa0 + (size_t)16 * kIP;
  const u16* Ast0 = stP + (size_t)(a * kB + b0) * kS;
  const u16* Ast1 = Ast0 + (size_t)16 * kS;
  u16* out1 = saEnc + (size_t)(a * kB + b0) * kH;
  u16* out2 = sEnc  + (size_t)(a * kB + b0) * kH;
  for (int ct = wave; ct < 16; ct += 8) {
    int n0 = ct * 16;
    // sa encoder, K = 160
    v8f acc0 = vzero(), acc1 = vzero();
    const u16* Wb = WencT + (size_t)a * kH * kIP + (size_t)n0 * kIP;
    for (int k0 = 0; k0 < kIP; k0 += 32) {
      V16 Bm = loadB(Wb, kIP, k0, lane);
      pfB(Wb, kIP, k0 + 32, lane);
      acc0 = bwmma(loadA(Asa0, kIP, k0, lane), Bm, acc0);
      acc1 = bwmma(loadA(Asa1, kIP, k0, lane), Bm, acc1);
    }
    float bias = b_enc[a * kH + n0 + n];
    for (int r = 0; r < 8; r++) {
      out1[(size_t)(hi * 8 + r) * kH + n0 + n]      = f2bf(lrelu(acc0[r] + bias));
      out1[(size_t)(16 + hi * 8 + r) * kH + n0 + n] = f2bf(lrelu(acc1[r] + bias));
    }
    // state encoder, K = 128
    acc0 = vzero(); acc1 = vzero();
    const u16* Ws = WsencT + (size_t)a * kH * kS + (size_t)n0 * kS;
    for (int k0 = 0; k0 < kS; k0 += 32) {
      V16 Bm = loadB(Ws, kS, k0, lane);
      pfB(Ws, kS, k0 + 32, lane);
      acc0 = bwmma(loadA(Ast0, kS, k0, lane), Bm, acc0);
      acc1 = bwmma(loadA(Ast1, kS, k0, lane), Bm, acc1);
    }
    bias = b_senc[a * kH + n0 + n];
    for (int r = 0; r < 8; r++) {
      out2[(size_t)(hi * 8 + r) * kH + n0 + n]      = f2bf(lrelu(acc0[r] + bias));
      out2[(size_t)(16 + hi * 8 + r) * kH + n0 + n] = f2bf(lrelu(acc1[r] + bias));
    }
  }
}

// -------- head projections (M=32 blocking): keys/sels/vals [head][agent][b][64] bf16 ----------
__global__ __launch_bounds__(256) void proj_kernel(
    const u16* __restrict__ saEnc, const u16* __restrict__ sEnc,
    const u16* __restrict__ WkT, const u16* __restrict__ WselT, const u16* __restrict__ WvT,
    const float* __restrict__ bv,
    u16* __restrict__ keysP, u16* __restrict__ selsP, u16* __restrict__ valsP) {
  int a = blockIdx.y, b0 = blockIdx.x * 32;
  int wave = threadIdx.x >> 5, lane = threadIdx.x & 31;
  int n = lane & 15, hi = lane >> 4;
  for (int t = wave; t < 48; t += 8) {
    int proj = t >> 4, rem = t & 15, head = rem >> 2, ct = rem & 3;
    const u16* Ab0 = (proj == 1 ? sEnc : saEnc) + (size_t)(a * kB + b0) * kH;
    const u16* Ab1 = Ab0 + (size_t)16 * kH;
    const u16* Wt = (proj == 0 ? WkT : proj == 1 ? WselT : WvT)
                    + (size_t)head * kD * kH + (size_t)(ct * 16) * kH;
    v8f acc0 = vzero(), acc1 = vzero();
    for (int k0 = 0; k0 < kH; k0 += 32) {
      V16 Bm = loadB(Wt, kH, k0, lane);
      pfB(Wt, kH, k0 + 32, lane);
      acc0 = bwmma(loadA(Ab0, kH, k0, lane), Bm, acc0);
      acc1 = bwmma(loadA(Ab1, kH, k0, lane), Bm, acc1);
    }
    u16* ob = (proj == 0 ? keysP : proj == 1 ? selsP : valsP);
    u16* out = ob + (((size_t)head * kN + a) * kB + b0) * kD + ct * 16;
    float bias = (proj == 2) ? bv[head * kD + ct * 16 + n] : 0.f;
    for (int r = 0; r < 8; r++) {
      float v0 = acc0[r], v1 = acc1[r];
      if (proj == 2) { v0 = lrelu(v0 + bias); v1 = lrelu(v1 + bias); }
      out[(size_t)(hi * 8 + r) * kD + n]      = f2bf(v0);
      out[(size_t)(16 + hi * 8 + r) * kD + n] = f2bf(v1);
    }
  }
}

// ---------------- attention: logits WMMA, softmax, other = probs@vals, reg accum --------------
__global__ __launch_bounds__(256) void attn_kernel(
    const u16* __restrict__ keysP, const u16* __restrict__ selsP, const u16* __restrict__ valsP,
    u16* __restrict__ otherP, float* __restrict__ regAcc) {
  __shared__ __align__(16) u16 valsT[8][kD * 16];   // per wave: [d][j]
  __shared__ __align__(16) u16 probsS[8][16 * 32];  // per wave: [i][j], j padded to 32
  int wave = threadIdx.x >> 5, lane = threadIdx.x & 31;
  int head = wave & 3;
  int b = blockIdx.x * 2 + (wave >> 2);
  int n = lane & 15, hi = lane >> 4;

  // stage vals tile transposed into LDS: valsT[d][j] = vals[head][j][b][d]
  u16* vT = valsT[wave];
  for (int idx = lane; idx < 128; idx += 32) {   // 16 rows x 8 chunks of 8 halves
    int j = idx >> 3, c = idx & 7;
    Q4H t; t.q = *(const uint4*)(valsP + (((size_t)head * kN + j) * kB + b) * kD + c * 8);
    for (int e = 0; e < 8; e++) vT[(c * 8 + e) * 16 + j] = t.h[e];
  }
  u16* pS = probsS[wave];
  for (int idx = lane; idx < 16 * 32; idx += 32) pS[idx] = 0;  // zero incl. K-pad

  // logits(16x16) = sels(16x64) @ keys^T(64x16): two K=32 WMMAs, contiguous loads
  const u16* selsBase = selsP + ((size_t)head * kN * kB + b) * kD;
  const u16* keysBase = keysP + ((size_t)head * kN * kB + b) * kD;
  v8f lg = vzero();
  for (int k0 = 0; k0 < kD; k0 += 32)
    lg = bwmma(loadA(selsBase, (size_t)kB * kD, k0, lane),
               loadB(keysBase, (size_t)kB * kD, k0, lane), lg);

  // regularizer: sum of raw off-diagonal logits^2, reduced over j (lanes) per row i
  for (int r = 0; r < 8; r++) {
    int i = hi * 8 + r;
    float v = (i == n) ? 0.f : lg[r] * lg[r];
    v += __shfl_xor(v, 1, 32); v += __shfl_xor(v, 2, 32);
    v += __shfl_xor(v, 4, 32); v += __shfl_xor(v, 8, 32);
    if (n == 0) atomicAdd(&regAcc[i], v);
  }

  // softmax over j (masked self, scaled by 1/sqrt(64)=0.125)
  const float NEG = -__builtin_huge_valf();
  for (int r = 0; r < 8; r++) {
    int i = hi * 8 + r;
    float s = (i == n) ? NEG : lg[r] * 0.125f;
    float m = s;
    m = fmaxf(m, __shfl_xor(m, 1, 32)); m = fmaxf(m, __shfl_xor(m, 2, 32));
    m = fmaxf(m, __shfl_xor(m, 4, 32)); m = fmaxf(m, __shfl_xor(m, 8, 32));
    float e = __expf(s - m);
    float sum = e;
    sum += __shfl_xor(sum, 1, 32); sum += __shfl_xor(sum, 2, 32);
    sum += __shfl_xor(sum, 4, 32); sum += __shfl_xor(sum, 8, 32);
    pS[i * 32 + n] = f2bf(e / sum);
  }

  // other(16x64) = probs(16x16, padded K=32) @ vals(16x64): 4 col-tile WMMAs
  for (int ct = 0; ct < 4; ct++) {
    V16 A = loadA(pS, 32, 0, lane);
    V16 Bm;
    if (hi == 0) {   // k=j=0..15 real
      const u16* p = vT + (ct * 16 + n) * 16;
      Bm.q[0] = *(const uint4*)p; Bm.q[1] = *(const uint4*)(p + 8);
    } else {         // k=j=16..31: zero pad
      uint4 z; z.x = z.y = z.z = z.w = 0u; Bm.q[0] = z; Bm.q[1] = z;
    }
    v8f acc = bwmma(A, Bm, vzero());
    for (int r = 0; r < 8; r++) {
      int i = hi * 8 + r;
      otherP[((size_t)i * kB + b) * kH + head * kD + ct * 16 + n] = f2bf(acc[r]);
    }
  }
}

// ------- critic (M=32 blocking): hdn/bh GEMMs -> LDS -> all_q / b -> argmax gather -> qb -----
__global__ __launch_bounds__(256) void critic_kernel(
    const u16* __restrict__ sEnc, const u16* __restrict__ otherP,
    const u16* __restrict__ Wc1T, const float* __restrict__ bc1,
    const u16* __restrict__ Wc2T, const float* __restrict__ bc2,
    const u16* __restrict__ Wb1T, const float* __restrict__ bb1,
    const u16* __restrict__ Wb2T, const float* __restrict__ bb2,
    const float* __restrict__ actions, float* __restrict__ out) {
  __shared__ __align__(16) u16 hdnS[32 * kH];
  __shared__ __align__(16) u16 bhS[32 * kH];
  __shared__ float allqS[32 * 16];
  __shared__ float bbS[32];
  int a = blockIdx.y, b0 = blockIdx.x * 32;
  int wave = threadIdx.x >> 5, lane = threadIdx.x & 31;
  int n = lane & 15, hi = lane >> 4;
  const u16* sE0 = sEnc   + (size_t)(a * kB + b0) * kH;
  const u16* sE1 = sE0 + (size_t)16 * kH;
  const u16* oE0 = otherP + (size_t)(a * kB + b0) * kH;
  const u16* oE1 = oE0 + (size_t)16 * kH;

  for (int t = wave; t < 32; t += 8) {
    if (t < 16) {  // hdn col-tile t, K=512 over [s_enc | other], both row tiles
      int n0 = t * 16;
      v8f acc0 = vzero(), acc1 = vzero();
      const u16* Wt = Wc1T + (size_t)a * kH * kH2 + (size_t)n0 * kH2;
      for (int k0 = 0; k0 < kH2; k0 += 32) {
        V16 Bm = loadB(Wt, kH2, k0, lane);
        pfB(Wt, kH2, k0 + 32, lane);
        const u16* A0 = (k0 < kH) ? sE0 : oE0;
        const u16* A1 = (k0 < kH) ? sE1 : oE1;
        int kk = (k0 < kH) ? k0 : k0 - kH;
        acc0 = bwmma(loadA(A0, kH, kk, lane), Bm, acc0);
        acc1 = bwmma(loadA(A1, kH, kk, lane), Bm, acc1);
      }
      float bias = bc1[a * kH + n0 + n];
      for (int r = 0; r < 8; r++) {
        hdnS[(hi * 8 + r) * kH + n0 + n]      = f2bf(lrelu(acc0[r] + bias));
        hdnS[(16 + hi * 8 + r) * kH + n0 + n] = f2bf(lrelu(acc1[r] + bias));
      }
    } else {       // bh col-tile, K=256 from s_enc, both row tiles
      int n0 = (t - 16) * 16;
      v8f acc0 = vzero(), acc1 = vzero();
      const u16* Wt = Wb1T + (size_t)a * kH * kH + (size_t)n0 * kH;
      for (int k0 = 0; k0 < kH; k0 += 32) {
        V16 Bm = loadB(Wt, kH, k0, lane);
        pfB(Wt, kH, k0 + 32, lane);
        acc0 = bwmma(loadA(sE0, kH, k0, lane), Bm, acc0);
        acc1 = bwmma(loadA(sE1, kH, k0, lane), Bm, acc1);
      }
      float bias = bb1[a * kH + n0 + n];
      for (int r = 0; r < 8; r++) {
        bhS[(hi * 8 + r) * kH + n0 + n]      = f2bf(lrelu(acc0[r] + bias));
        bhS[(16 + hi * 8 + r) * kH + n0 + n] = f2bf(lrelu(acc1[r] + bias));
      }
    }
  }
  __syncthreads();
  if (wave < 2) {        // all_q = hdn @ Wc2 (padded 16 cols; cols>=8 zero-weight)
    const u16* Ah = hdnS + (size_t)(wave * 16) * kH;
    v8f acc = vzero();
    const u16* Wt = Wc2T + (size_t)a * 16 * kH;
    for (int k0 = 0; k0 < kH; k0 += 32)
      acc = bwmma(loadA(Ah, kH, k0, lane), loadB(Wt, kH, k0, lane), acc);
    for (int r = 0; r < 8; r++) allqS[(wave * 16 + hi * 8 + r) * 16 + n] = acc[r];
  } else if (wave < 4) { // b = bh @ Wb2 (padded; only col 0 valid)
    int rt = wave - 2;
    const u16* Ah = bhS + (size_t)(rt * 16) * kH;
    v8f acc = vzero();
    const u16* Wt = Wb2T + (size_t)a * 16 * kH;
    for (int k0 = 0; k0 < kH; k0 += 32)
      acc = bwmma(loadA(Ah, kH, k0, lane), loadB(Wt, kH, k0, lane), acc);
    if (n == 0) for (int r = 0; r < 8; r++) bbS[rt * 16 + hi * 8 + r] = acc[r];
  }
  __syncthreads();
  if (threadIdx.x < 32) {
    int m = threadIdx.x;
    const float* act = actions + (size_t)(a * kB + b0 + m) * kA;
    int amax = 0; float best = act[0];
    for (int j = 1; j < kA; j++) { float v = act[j]; if (v > best) { best = v; amax = j; } }
    float q  = allqS[m * 16 + amax] + bc2[a * kA + amax];
    float bb = bbS[m] + bb2[a];
    out[(size_t)a * kB + b0 + m] = q - bb;
  }
}

__global__ void reg_kernel(const float* __restrict__ regAcc, float* __restrict__ out) {
  int i = threadIdx.x;
  if (i < kN) out[(size_t)kN * kB + i] = 0.001f * regAcc[i] / (float)(kB * (kN - 1));
}

// ------------------------------------ host launcher ------------------------------------------
extern "C" void kernel_launch(void* const* d_in, const int* in_sizes, int n_in,
                              void* d_out, int out_size, void* d_ws, size_t ws_size,
                              hipStream_t stream) {
  const float* states  = (const float*)d_in[0];
  const float* actions = (const float*)d_in[1];
  const float* sa      = (const float*)d_in[2];
  const float* W_enc   = (const float*)d_in[3];
  const float* b_enc   = (const float*)d_in[4];
  const float* W_senc  = (const float*)d_in[5];
  const float* b_senc  = (const float*)d_in[6];
  const float* Wk      = (const float*)d_in[7];
  const float* Wsel    = (const float*)d_in[8];
  const float* Wv      = (const float*)d_in[9];
  const float* bv      = (const float*)d_in[10];
  const float* Wc1     = (const float*)d_in[11];
  const float* bc1     = (const float*)d_in[12];
  const float* Wc2     = (const float*)d_in[13];
  const float* bc2     = (const float*)d_in[14];
  const float* Wb1     = (const float*)d_in[15];
  const float* bb1     = (const float*)d_in[16];
  const float* Wb2     = (const float*)d_in[17];
  const float* bb2     = (const float*)d_in[18];
  float* out = (float*)d_out;

  u16* W = (u16*)d_ws;
  size_t off = 0;
  auto take = [&](size_t elems) { size_t r = off; off += elems; off = (off + 127) & ~(size_t)127; return r; };

  float* regAcc = (float*)(W + take(32));                         // 16 f32
  u16* saP    = W + take((size_t)kN * kB * kIP);
  u16* stP    = W + take((size_t)kN * kB * kS);
  u16* WencT  = W + take((size_t)kN * kH * kIP);
  u16* WsencT = W + take((size_t)kN * kH * kS);
  u16* WkT    = W + take((size_t)kHeads * kD * kH);
  u16* WselT  = W + take((size_t)kHeads * kD * kH);
  u16* WvT    = W + take((size_t)kHeads * kD * kH);
  u16* Wc1T   = W + take((size_t)kN * kH * kH2);
  u16* Wc2T   = W + take((size_t)kN * 16 * kH);
  u16* Wb1T   = W + take((size_t)kN * kH * kH);
  u16* Wb2T   = W + take((size_t)kN * 16 * kH);
  u16* saEnc  = W + take((size_t)kN * kB * kH);
  u16* sEnc   = W + take((size_t)kN * kB * kH);
  u16* keysP  = W + take((size_t)kHeads * kN * kB * kD);
  u16* selsP  = W + take((size_t)kHeads * kN * kB * kD);
  u16* valsP  = W + take((size_t)kHeads * kN * kB * kD);
  u16* otherP = W + take((size_t)kN * kB * kH);
  (void)ws_size; (void)in_sizes; (void)n_in; (void)out_size;

  // weight prep: src[K][Nc] -> dst[Npad][Kpad] (bf16, transposed, zero-padded)
  wprep<<<512, 256, 0, stream>>>(W_enc,  WencT,  kN, kI,  kH,  kIP, kH);
  wprep<<<512, 256, 0, stream>>>(W_senc, WsencT, kN, kS,  kH,  kS,  kH);
  wprep<<<64,  256, 0, stream>>>(Wk,   WkT,   kHeads, kH, kD, kH, kD);
  wprep<<<64,  256, 0, stream>>>(Wsel, WselT, kHeads, kH, kD, kH, kD);
  wprep<<<64,  256, 0, stream>>>(Wv,   WvT,   kHeads, kH, kD, kH, kD);
  wprep<<<1024, 256, 0, stream>>>(Wc1, Wc1T, kN, kH2, kH, kH2, kH);
  wprep<<<64,  256, 0, stream>>>(Wc2, Wc2T, kN, kH, kA, kH, 16);
  wprep<<<512, 256, 0, stream>>>(Wb1, Wb1T, kN, kH, kH, kH, kH);
  wprep<<<64,  256, 0, stream>>>(Wb2, Wb2T, kN, kH, 1,  kH, 16);
  aprep<<<2048, 256, 0, stream>>>(sa, saP, states, stP, regAcc);

  enc_kernel<<<dim3(kB / 32, kN), 256, 0, stream>>>(saP, stP, WencT, b_enc, WsencT, b_senc, saEnc, sEnc);
  proj_kernel<<<dim3(kB / 32, kN), 256, 0, stream>>>(saEnc, sEnc, WkT, WselT, WvT, bv, keysP, selsP, valsP);
  attn_kernel<<<dim3(kB / 2), 256, 0, stream>>>(keysP, selsP, valsP, otherP, regAcc);
  critic_kernel<<<dim3(kB / 32, kN), 256, 0, stream>>>(sEnc, otherP, Wc1T, bc1, Wc2T, bc2,
                                                       Wb1T, bb1, Wb2T, bb2, actions, out);
  reg_kernel<<<1, 32, 0, stream>>>(regAcc, out);
}
